// VQEmbedding_Original_3624952398142
// MI455X (gfx1250) — matrix operand order repaired and checked by reading, not compiled, not run on hardware
//
#include <hip/hip_runtime.h>
#include <stdint.h>

// VQ nearest-codebook search on gfx1250 (MI455X).
//   argmin_n ||z - w_n||^2  ==  argmax_n (z.w_n - 0.5*||w_n||^2)
// GEMM via bf16x3 split WMMA (V_WMMA_F32_16X16X32_BF16), LDS double-buffered
// with GLOBAL_LOAD_ASYNC_TO_LDS_B128 (ASYNCcnt) so the L2->LDS tile fill
// overlaps the matrix work.

typedef __attribute__((ext_vector_type(16))) __bf16        v16bf;
typedef __attribute__((ext_vector_type(8)))  float         v8f;
typedef __attribute__((ext_vector_type(4)))  float         v4f;
typedef __attribute__((ext_vector_type(4)))  unsigned int  v4u;
typedef __attribute__((ext_vector_type(8)))  unsigned int  v8u;

#define KCB  1024  // codebook size
#define DIM  256   // embedding dim
#define HWSZ 1024  // H*W (feature stride in NCHW)

// LDS / swizzled-W image geometry (dword units)
#define B_STRIDE 12              // 8 fragment dwords + 4 pad per lane (bank-conflict-free b128)
#define B_KB     (B_STRIDE * 32) // 384 dwords per k-block (32 lanes)
#define B_MAT    (B_KB * 8)      // 3072 dwords per matrix (hi or lo), 8 k-blocks
#define TILE_DW  (2 * B_MAT)     // 6144 dwords per 16-column tile (hi + lo)

__device__ __forceinline__ unsigned int bf16_rne(float f) {
    unsigned int u = __float_as_uint(f);
    return (u + 0x7FFFu + ((u >> 16) & 1u)) >> 16;  // round-to-nearest-even
}
__device__ __forceinline__ float bf16_tof(unsigned int h) {
    return __uint_as_float(h << 16);
}

// ---------- Pass 1: wsq[k] = ||W_k||^2 (exact fp32) ----------
__global__ __launch_bounds__(256) void vq_wsq_kernel(const float* __restrict__ W,
                                                     float* __restrict__ wsq) {
    int n = blockIdx.x * blockDim.x + threadIdx.x;
    if (n < KCB) {
        const float* row = W + n * DIM;
        float s = 0.f;
#pragma unroll 8
        for (int i = 0; i < DIM; i += 4) {
            v4f v = *reinterpret_cast<const v4f*>(row + i);
            s += v.x * v.x + v.y * v.y + v.z * v.z + v.w * v.w;
        }
        wsq[n] = s;
    }
}

// ---------- Pass 2: split W into bf16 hi/lo, swizzled into B-fragment image ----------
// Tile t, k-block kb, lane l = n + 16*kh, fragment dword j holds the bf16 pair
// W[t*16+n][c], W[..][c+1] with c = kb*32 + kh*16 + 2*j.
__global__ __launch_bounds__(256) void vq_wswz_kernel(const float* __restrict__ W,
                                                      unsigned int* __restrict__ wswz) {
    const int t   = blockIdx.x;
    const int tid = threadIdx.x;
    const int n   = tid >> 4;
    const int g   = tid & 15;
    const int kb  = g >> 1;
    const int kh  = g & 1;
    const int l   = n + 16 * kh;

    const float* src = W + (size_t)(t * 16 + n) * DIM + g * 16;  // 16 contiguous floats
    float w[16];
#pragma unroll
    for (int q = 0; q < 4; ++q) {
        v4f v = *reinterpret_cast<const v4f*>(src + 4 * q);
        w[4 * q + 0] = v.x; w[4 * q + 1] = v.y; w[4 * q + 2] = v.z; w[4 * q + 3] = v.w;
    }
    unsigned int hi[8], lo[8];
#pragma unroll
    for (int p = 0; p < 8; ++p) {
        const float f0 = w[2 * p], f1 = w[2 * p + 1];
        const unsigned int h0 = bf16_rne(f0), h1 = bf16_rne(f1);
        const float l0 = f0 - bf16_tof(h0), l1 = f1 - bf16_tof(h1);
        hi[p] = h0 | (h1 << 16);
        lo[p] = bf16_rne(l0) | (bf16_rne(l1) << 16);
    }
    unsigned int* dh = wswz + (size_t)t * TILE_DW + kb * B_KB + l * B_STRIDE;
    unsigned int* dl = dh + B_MAT;
    *reinterpret_cast<v4u*>(dh)     = (v4u){hi[0], hi[1], hi[2], hi[3]};
    *reinterpret_cast<v4u*>(dh + 4) = (v4u){hi[4], hi[5], hi[6], hi[7]};
    *reinterpret_cast<v4u*>(dl)     = (v4u){lo[0], lo[1], lo[2], lo[3]};
    *reinterpret_cast<v4u*>(dl + 4) = (v4u){lo[4], lo[5], lo[6], lo[7]};
}

// ---------- Pass 3: fused bf16x3 WMMA GEMM + argmax ----------
// 256 blocks x 256 threads (8 waves). Wave w of block b owns rows
// [b*128 + w*16, +16). Row r: batch = r>>10, hw = r&1023,
// z(r, c) = z[batch*DIM*HWSZ + c*HWSZ + hw].
__global__ __launch_bounds__(256, 1) void vq_argmin_kernel(const float* __restrict__ z,
                                                           const unsigned int* __restrict__ wswz,
                                                           const float* __restrict__ wsq,
                                                           int* __restrict__ out) {
    __shared__ unsigned int ldsb[2 * TILE_DW];  // double-buffered tile (48 KB)

    const int tid   = threadIdx.x;
    const int wave  = tid >> 5;
    const int lane  = tid & 31;
    const int mlane = lane & 15;  // A: row-in-tile, B/C: column-in-tile
    const int khalf = lane >> 4;

    const int r0 = blockIdx.x * 128 + wave * 16;
    const int r  = r0 + mlane;
    const int bb = r >> 10;
    const int hw = r & 1023;
    const float* pz = z + (size_t)bb * (DIM * HWSZ) + hw;

    // AS3 byte offset of the LDS buffer (generic shared-pointer low 32 bits).
    const unsigned int ldsbase = (unsigned int)(uintptr_t)(&ldsb[0]);

    // Async fill of one tile image (6144 dwords) into LDS buffer `buf`.
    // 6 x b128 per thread, fully coalesced from L2, tracked with ASYNCcnt.
    auto fill_async = [&](int tile, int buf) {
        const unsigned int* src = wswz + (size_t)tile * TILE_DW;
#pragma unroll
        for (int c = 0; c < 6; ++c) {
            const int o = c * 1024 + tid * 4;  // dword offset within tile image
            const unsigned int* gp = src + o;
            const unsigned int  la = ldsbase + (unsigned int)(buf * TILE_DW + o) * 4u;
            asm volatile("global_load_async_to_lds_b128 %0, %1, off"
                         :: "v"(la), "v"(gp) : "memory");
        }
    };

    // Preload A in the 16-bit 16x32 WMMA layout, split into bf16 hi/lo.
    // Fragment VGPR v: K = kh*16 + 2v (contiguous, matches swizzled B).
    v8u ahiv[8], alov[8];
#pragma unroll
    for (int kb = 0; kb < 8; ++kb) {
#pragma unroll
        for (int v = 0; v < 8; ++v) {
            const int c = kb * 32 + khalf * 16 + 2 * v;
            const float f0 = pz[c * HWSZ];
            const float f1 = pz[(c + 1) * HWSZ];
            const unsigned int h0 = bf16_rne(f0), h1 = bf16_rne(f1);
            const float l0 = f0 - bf16_tof(h0), l1 = f1 - bf16_tof(h1);
            ahiv[kb][v] = h0 | (h1 << 16);
            alov[kb][v] = bf16_rne(l0) | (bf16_rne(l1) << 16);
        }
    }

    float best[8];
    int   bidx[8];
#pragma unroll
    for (int j = 0; j < 8; ++j) { best[j] = -3.402823466e38f; bidx[j] = 0; }

    // Prologue: stage tile 0 into buffer 0.
    fill_async(0, 0);
    asm volatile("s_wait_asynccnt 0x0" ::: "memory");
    __syncthreads();

    for (int t = 0; t < 64; ++t) {  // 64 column tiles of 16 codebook entries
        // Kick off next tile's fill into the other buffer (prev readers of that
        // buffer finished at the end-of-(t-1) barrier).
        if (t < 63) fill_async(t + 1, (t + 1) & 1);

        const unsigned int* bph = ldsb + (t & 1) * TILE_DW + lane * B_STRIDE;  // hi
        const unsigned int* bpl = bph + B_MAT;                                 // lo

        v8f acc = {};
#pragma unroll
        for (int kb = 0; kb < 8; ++kb) {
            const v4u bh0 = *reinterpret_cast<const v4u*>(bph + kb * B_KB);
            const v4u bh1 = *reinterpret_cast<const v4u*>(bph + kb * B_KB + 4);
            const v4u bl0 = *reinterpret_cast<const v4u*>(bpl + kb * B_KB);
            const v4u bl1 = *reinterpret_cast<const v4u*>(bpl + kb * B_KB + 4);
            const v16bf bh = __builtin_bit_cast(v16bf,
                __builtin_shufflevector(bh0, bh1, 0, 1, 2, 3, 4, 5, 6, 7));
            const v16bf bl = __builtin_bit_cast(v16bf,
                __builtin_shufflevector(bl0, bl1, 0, 1, 2, 3, 4, 5, 6, 7));
            const v16bf ah = __builtin_bit_cast(v16bf, ahiv[kb]);
            const v16bf al = __builtin_bit_cast(v16bf, alov[kb]);
            acc = __builtin_amdgcn_wmma_f32_16x16x32_bf16(false, ah, false, bh,
                                                          (short)0, acc, false, false);
            acc = __builtin_amdgcn_wmma_f32_16x16x32_bf16(false, al, false, bh,
                                                          (short)0, acc, false, false);
            acc = __builtin_amdgcn_wmma_f32_16x16x32_bf16(false, ah, false, bl,
                                                          (short)0, acc, false, false);
        }

        // score = z.w - 0.5*||w||^2 ; strict > keeps jnp.argmin's first-min ties
        const int   n0   = t * 16;
        const float wn   = wsq[n0 + mlane];
        const int   ncol = n0 + mlane;
#pragma unroll
        for (int j = 0; j < 8; ++j) {
            const float s = acc[j] - 0.5f * wn;
            if (s > best[j]) { best[j] = s; bidx[j] = ncol; }
        }

        // Next tile's async fill must be LDS-visible to all waves.
        asm volatile("s_wait_asynccnt 0x0" ::: "memory");
        __syncthreads();
    }

    // Argmax across the 16 lanes of each half-wave (lane = N residue,
    // VGPR j = row m = j + 8*khalf). Ties -> smaller codebook index.
#pragma unroll
    for (int j = 0; j < 8; ++j) {
#pragma unroll
        for (int off = 1; off < 16; off <<= 1) {
            const float os = __shfl_xor(best[j], off, 32);
            const int   oi = __shfl_xor(bidx[j], off, 32);
            if (os > best[j] || (os == best[j] && oi < bidx[j])) {
                best[j] = os;
                bidx[j] = oi;
            }
        }
    }

    if (mlane == 0) {
#pragma unroll
        for (int j = 0; j < 8; ++j)
            out[r0 + j + 8 * khalf] = bidx[j];
    }
}

extern "C" void kernel_launch(void* const* d_in, const int* in_sizes, int n_in,
                              void* d_out, int out_size, void* d_ws, size_t ws_size,
                              hipStream_t stream) {
    const float* z = (const float*)d_in[0];  // [32, 256, 32, 32] fp32 NCHW
    const float* W = (const float*)d_in[1];  // [1024, 256] fp32
    int*         out = (int*)d_out;          // [32768] int32 indices

    // Workspace layout: [64 tiles * 6144 dwords swizzled bf16 hi/lo W][1024 f32 wsq]
    unsigned int* wswz = (unsigned int*)d_ws;
    float*        wsq  = (float*)((char*)d_ws + (size_t)64 * TILE_DW * 4);

    vq_wsq_kernel<<<dim3(4), dim3(256), 0, stream>>>(W, wsq);
    vq_wswz_kernel<<<dim3(64), dim3(256), 0, stream>>>(W, wswz);
    vq_argmin_kernel<<<dim3(256), dim3(256), 0, stream>>>(z, wswz, wsq, out);
}
